// BERTLSTMClassification_74105365725705
// MI455X (gfx1250) — matrix operand order repaired and checked by reading, not compiled, not run on hardware
//
#include <hip/hip_runtime.h>
#include <hip/hip_bf16.h>

// ---------------------------------------------------------------------------
// BERT-pool + BiLSTM + aspect classification for MI455X (gfx1250, wave32).
// GEMM stages (input gates, recurrence, FC1) run on v_wmma_f32_16x16x32_f16
// with 1x4 register blocking (A fragment reused across 4 N-tiles).
// ---------------------------------------------------------------------------

#define BATCH    64
#define S_WP     512
#define D_BERT   768
#define NWORDS   256
#define HID      256
#define G4       1024      // 4*HID
#define H2       512       // 2*HID
#define A_MAX    8
#define K_MAX    4
#define N_LABELS 4

typedef __attribute__((ext_vector_type(16))) _Float16 v16h;
typedef __attribute__((ext_vector_type(8)))  float    v8f;

static __device__ __forceinline__ v8f wmma_f16(v16h a, v16h b, v8f c) {
  return __builtin_amdgcn_wmma_f32_16x16x32_f16(
      /*neg_a=*/false, a, /*neg_b=*/false, b,
      /*c_mod=*/(short)0, c, /*reuse_a=*/false, /*reuse_b=*/false);
}

// A fragment: 16x32 f16 tile, row-major source with row stride ldk (halves).
// Lane L (row = L&15) holds K = {c*8+j, 16+c*8+j}, c = L>>4.  (ISA 7.12.2)
static __device__ __forceinline__ v16h load_a_frag(const _Float16* A, int ldk,
                                                   int m0, int k0, int lane) {
  const int row = m0 + (lane & 15);
  const int c   = lane >> 4;
  const _Float16* p = A + (size_t)row * ldk + k0;
  v16h a;
#pragma unroll
  for (int j = 0; j < 8; ++j) {
    a[j]     = p[c * 8 + j];
    a[8 + j] = p[16 + c * 8 + j];
  }
  return a;
}

// B fragment: 32x16 (KxN) from an [N][K] row-major weight array (w[g][d]).
// Lane L holds column n = L&15, K = {c*8+j, 16+c*8+j}, c = L>>4 (same as A).
static __device__ __forceinline__ v16h load_b_frag(const _Float16* W, int ldk,
                                                   int n0, int k0, int lane) {
  const int n    = n0 + (lane & 15);
  const int koff = (lane >> 4) * 8;
  const _Float16* p = W + (size_t)n * ldk + k0 + koff;
  v16h b;
#pragma unroll
  for (int j = 0; j < 8; ++j) { b[j] = p[j]; b[8 + j] = p[16 + j]; }
  return b;
}

static __device__ __forceinline__ float sigmoidf(float x) {
  return 1.0f / (1.0f + expf(-x));
}

// ---------------------------------------------------------------------------
// K1: wordpiece -> word segment mean.  mask is sorted per row, so each word's
// wordpieces are a contiguous [lo,hi) range found by binary search.
// grid (NWORDS, BATCH), block 256.
// ---------------------------------------------------------------------------
__global__ void seg_mean_kernel(const float* __restrict__ emb,
                                const int* __restrict__ mask,
                                _Float16* __restrict__ words) {
  const int w = blockIdx.x;
  const int b = blockIdx.y;
  const int* m = mask + (size_t)b * S_WP;

  int l = 0, r = S_WP;                       // lower_bound(w)
  while (l < r) { int mid = (l + r) >> 1; if (m[mid] < w) l = mid + 1; else r = mid; }
  const int lo = l;
  r = S_WP;                                  // lower_bound(w+1)
  while (l < r) { int mid = (l + r) >> 1; if (m[mid] < w + 1) l = mid + 1; else r = mid; }
  const int hi = l;

  const float inv = 1.0f / fmaxf((float)(hi - lo), 1.0f);
  for (int d = threadIdx.x; d < D_BERT; d += blockDim.x) {
    float s = 0.0f;
    for (int t = lo; t < hi; ++t)
      s += emb[((size_t)b * S_WP + t) * D_BERT + d];
    words[((size_t)b * NWORDS + w) * D_BERT + d] = (_Float16)(s * inv);
  }
}

// ---------------------------------------------------------------------------
// K2: f32 -> f16 weight conversion.
// ---------------------------------------------------------------------------
__global__ void cvt_f16_kernel(const float* __restrict__ in,
                               _Float16* __restrict__ out, int n) {
  int i = blockIdx.x * blockDim.x + threadIdx.x;
  if (i < n) out[i] = (_Float16)in[i];
}

// ---------------------------------------------------------------------------
// K3: xg = words @ w_ih^T + b_ih + b_hh, both directions (blockIdx.z).
// M = B*NWORDS = 16384, K = 768, N = 1024.
// Each wave computes a 16x64 strip: A fragment reused across 4 WMMAs.
// grid (2, 1024, 2), block 256 (8 waves).
// ---------------------------------------------------------------------------
__global__ void xg_gemm_kernel(const _Float16* __restrict__ words,
                               const _Float16* __restrict__ wih_f,
                               const _Float16* __restrict__ wih_r,
                               const float* __restrict__ bih_f,
                               const float* __restrict__ bhh_f,
                               const float* __restrict__ bih_r,
                               const float* __restrict__ bhh_r,
                               float* __restrict__ xg_f,
                               float* __restrict__ xg_r) {
  const int wave = threadIdx.x >> 5, lane = threadIdx.x & 31;
  const int n0 = (blockIdx.x * 8 + wave) * 64;   // 64-wide N strip
  const int m0 = blockIdx.y * 16;
  const _Float16* W  = blockIdx.z ? wih_r : wih_f;
  const float*    bi = blockIdx.z ? bih_r : bih_f;
  const float*    bh = blockIdx.z ? bhh_r : bhh_f;
  float*          xg = blockIdx.z ? xg_r  : xg_f;

  v8f acc[4] = {{}, {}, {}, {}};
#pragma unroll 2
  for (int k0 = 0; k0 < D_BERT; k0 += 32) {
    const v16h a = load_a_frag(words, D_BERT, m0, k0, lane);
#pragma unroll
    for (int nt = 0; nt < 4; ++nt) {
      v16h b = load_b_frag(W, D_BERT, n0 + nt * 16, k0, lane);
      acc[nt] = wmma_f16(a, b, acc[nt]);
    }
  }
  const int n = lane & 15, mb = (lane >> 4) * 8;
#pragma unroll
  for (int nt = 0; nt < 4; ++nt) {
    const int nc = n0 + nt * 16 + n;
    const float bias = bi[nc] + bh[nc];
#pragma unroll
    for (int r = 0; r < 8; ++r)
      xg[((size_t)(m0 + mb + r)) * G4 + nc] = acc[nt][r] + bias;
  }
}

// ---------------------------------------------------------------------------
// K4: LSTM recurrence.  16 batches per workgroup, 8 waves x 128 gate columns.
// Per step: gates = xg[t] + h @ w_hh^T (WMMA, K=256).  The full 16x256 h tile
// (8 A fragments, 64 VGPRs) is loaded from LDS once per step and reused
// across all 8 N-tiles.  h kept f16 in LDS, c in registers (thread owns
// h_idx = tid for all 16 batches).  grid (BATCH/16, 2), block 256,
// dynamic LDS 73728 B.
// ---------------------------------------------------------------------------
__global__ void lstm_kernel(const float* __restrict__ xg_f,
                            const float* __restrict__ xg_r,
                            const _Float16* __restrict__ whh_f,
                            const _Float16* __restrict__ whh_r,
                            float* __restrict__ out) {
  extern __shared__ char smem_raw[];
  float*    gates = (float*)smem_raw;                       // [16][1024] 64 KB
  _Float16* hbuf  = (_Float16*)(smem_raw + 16 * G4 * 4);    // [16][256]   8 KB

  const int dir = blockIdx.y;
  const int b0  = blockIdx.x * 16;
  const float*    xg  = dir ? xg_r  : xg_f;
  const _Float16* whh = dir ? whh_r : whh_f;

  const int tid = threadIdx.x, wave = tid >> 5, lane = tid & 31;
  const int n_base = wave * 128;
  const int n = lane & 15, mb = (lane >> 4) * 8;

  float c[16];
#pragma unroll
  for (int j = 0; j < 16; ++j) c[j] = 0.0f;
#pragma unroll
  for (int j = 0; j < 16; ++j) hbuf[j * HID + tid] = (_Float16)0.0f;
  __syncthreads();

  for (int step = 0; step < NWORDS; ++step) {
    const int t = dir ? (NWORDS - 1 - step) : step;

    // --- load the whole 16x256 h tile once per step (8 A fragments) -------
    v16h afr[8];
#pragma unroll
    for (int kk = 0; kk < 8; ++kk)
      afr[kk] = load_a_frag(hbuf, HID, 0, kk * 32, lane);

    // --- WMMA phase: this wave's 8 gate tiles, acc seeded from xg[t] ------
    for (int nt = 0; nt < 8; ++nt) {
      const int n0 = n_base + nt * 16;
      v8f acc;
#pragma unroll
      for (int r = 0; r < 8; ++r)
        acc[r] = xg[(((size_t)(b0 + mb + r)) * NWORDS + t) * G4 + n0 + n];
#pragma unroll
      for (int kk = 0; kk < 8; ++kk) {
        v16h b = load_b_frag(whh, HID, n0, kk * 32, lane);
        acc = wmma_f16(afr[kk], b, acc);
      }
#pragma unroll
      for (int r = 0; r < 8; ++r)
        gates[(mb + r) * G4 + n0 + n] = acc[r];
    }
    __syncthreads();

    // --- elementwise cell update (thread owns h_idx = tid, 16 batches) ----
#pragma unroll
    for (int j = 0; j < 16; ++j) {
      const float gi = gates[j * G4 + tid];
      const float gf = gates[j * G4 + HID + tid];
      const float gg = gates[j * G4 + 2 * HID + tid];
      const float go = gates[j * G4 + 3 * HID + tid];
      c[j] = sigmoidf(gf) * c[j] + sigmoidf(gi) * tanhf(gg);
      const float h = sigmoidf(go) * tanhf(c[j]);
      hbuf[j * HID + tid] = (_Float16)h;
      out[(((size_t)(b0 + j)) * NWORDS + t) * H2 + dir * HID + tid] = h;
    }
    __syncthreads();
  }
}

// ---------------------------------------------------------------------------
// K5: sentence mean over the 256 padded word slots.  One thread per (b, d).
// ---------------------------------------------------------------------------
__global__ void sent_mean_kernel(const float* __restrict__ lstm_out,
                                 float* __restrict__ sent) {
  int i = blockIdx.x * blockDim.x + threadIdx.x;
  if (i >= BATCH * H2) return;
  const int b = i >> 9, d = i & (H2 - 1);
  float s = 0.0f;
  for (int t = 0; t < NWORDS; ++t)
    s += lstm_out[((size_t)b * NWORDS + t) * H2 + d];
  sent[i] = s * (1.0f / (float)NWORDS);
}

// ---------------------------------------------------------------------------
// K6: emb[b,a,:] = valid ? concat(sent_mean[b], asp_mean[b,a]) : 0   (f16)
// One thread per (b, a, d).
// ---------------------------------------------------------------------------
__global__ void build_emb_kernel(const float* __restrict__ lstm_out,
                                 const float* __restrict__ sent,
                                 const int* __restrict__ aidx,
                                 _Float16* __restrict__ emb) {
  int i = blockIdx.x * blockDim.x + threadIdx.x;
  if (i >= BATCH * A_MAX * 2 * H2) return;
  const int d  = i & (2 * H2 - 1);
  const int ba = i >> 10;
  const int b  = ba >> 3, a = ba & 7;
  const int* idx = aidx + (b * A_MAX + a) * K_MAX;

  int ntok = 0;
#pragma unroll
  for (int k = 0; k < K_MAX; ++k) ntok += (idx[k] >= 0);

  float val = 0.0f;
  if (ntok > 0) {
    if (d < H2) {
      val = sent[b * H2 + d];
    } else {
      const int dd = d - H2;
      float s = 0.0f;
#pragma unroll
      for (int k = 0; k < K_MAX; ++k) {
        if (idx[k] >= 0) {
          int t = idx[k];
          t = t < 0 ? 0 : (t > NWORDS - 2 ? NWORDS - 2 : t);
          s += lstm_out[((size_t)b * NWORDS + (t + 1)) * H2 + dd];
        }
      }
      val = s / (float)ntok;
    }
  }
  emb[i] = (_Float16)val;
}

// ---------------------------------------------------------------------------
// K7: FC1 via WMMA + bias + ReLU.  M=512, K=1024, N=256.
// Each wave computes a 16x64 strip (1x4 blocking); block covers 2 M-tiles.
// grid (1, 16), block 256 (8 waves).
// ---------------------------------------------------------------------------
__global__ void fc1_gemm_kernel(const _Float16* __restrict__ emb,
                                const _Float16* __restrict__ w1,
                                const float* __restrict__ b1,
                                float* __restrict__ h1) {
  const int wave = threadIdx.x >> 5, lane = threadIdx.x & 31;
  const int m0 = (blockIdx.y * 2 + (wave >> 2)) * 16;
  const int n0 = (wave & 3) * 64;

  v8f acc[4] = {{}, {}, {}, {}};
#pragma unroll 2
  for (int k0 = 0; k0 < 2 * H2; k0 += 32) {
    const v16h a = load_a_frag(emb, 2 * H2, m0, k0, lane);
#pragma unroll
    for (int nt = 0; nt < 4; ++nt) {
      v16h b = load_b_frag(w1, 2 * H2, n0 + nt * 16, k0, lane);
      acc[nt] = wmma_f16(a, b, acc[nt]);
    }
  }
  const int n = lane & 15, mb = (lane >> 4) * 8;
#pragma unroll
  for (int nt = 0; nt < 4; ++nt) {
    const int nc = n0 + nt * 16 + n;
    const float bias = b1[nc];
#pragma unroll
    for (int r = 0; r < 8; ++r)
      h1[(size_t)(m0 + mb + r) * HID + nc] = fmaxf(acc[nt][r] + bias, 0.0f);
  }
}

// ---------------------------------------------------------------------------
// K8: FC2 (tiny: 512x4 dots of length 256) -> logits.
// ---------------------------------------------------------------------------
__global__ void fc2_kernel(const float* __restrict__ h1,
                           const float* __restrict__ w2,
                           const float* __restrict__ b2,
                           float* __restrict__ out) {
  int i = blockIdx.x * blockDim.x + threadIdx.x;
  if (i >= BATCH * A_MAX * N_LABELS) return;
  const int row = i >> 2, l = i & 3;
  float s = b2[l];
  for (int k = 0; k < HID; ++k)
    s += h1[(size_t)row * HID + k] * w2[l * HID + k];
  out[i] = s;
}

// ---------------------------------------------------------------------------
extern "C" void kernel_launch(void* const* d_in, const int* in_sizes, int n_in,
                              void* d_out, int out_size, void* d_ws, size_t ws_size,
                              hipStream_t stream) {
  const float* embeddings = (const float*)d_in[0];
  const int*   mask       = (const int*)d_in[1];
  const int*   aidx       = (const int*)d_in[2];
  const float* w_ih_f     = (const float*)d_in[3];
  const float* w_hh_f     = (const float*)d_in[4];
  const float* b_ih_f     = (const float*)d_in[5];
  const float* b_hh_f     = (const float*)d_in[6];
  const float* w_ih_r     = (const float*)d_in[7];
  const float* w_hh_r     = (const float*)d_in[8];
  const float* b_ih_r     = (const float*)d_in[9];
  const float* b_hh_r     = (const float*)d_in[10];
  const float* fc1_w      = (const float*)d_in[11];
  const float* fc1_b      = (const float*)d_in[12];
  const float* fc2_w      = (const float*)d_in[13];
  const float* fc2_b      = (const float*)d_in[14];
  float* out = (float*)d_out;

  // ---- workspace carve-up (~191 MiB) ----
  char* ws = (char*)d_ws;
  size_t off = 0;
  auto alloc = [&](size_t bytes) -> void* {
    void* p = ws + off;
    off += (bytes + 255) & ~(size_t)255;
    return p;
  };
  _Float16* words_h = (_Float16*)alloc((size_t)BATCH * NWORDS * D_BERT * 2);
  _Float16* wihf_h  = (_Float16*)alloc((size_t)G4 * D_BERT * 2);
  _Float16* wihr_h  = (_Float16*)alloc((size_t)G4 * D_BERT * 2);
  _Float16* whhf_h  = (_Float16*)alloc((size_t)G4 * HID * 2);
  _Float16* whhr_h  = (_Float16*)alloc((size_t)G4 * HID * 2);
  _Float16* fc1w_h  = (_Float16*)alloc((size_t)HID * 2 * H2 * 2);
  float*    xg_f    = (float*)alloc((size_t)BATCH * NWORDS * G4 * 4);
  float*    xg_r    = (float*)alloc((size_t)BATCH * NWORDS * G4 * 4);
  float*    lstm_o  = (float*)alloc((size_t)BATCH * NWORDS * H2 * 4);
  float*    sent    = (float*)alloc((size_t)BATCH * H2 * 4);
  _Float16* emb_h   = (_Float16*)alloc((size_t)BATCH * A_MAX * 2 * H2 * 2);
  float*    h1      = (float*)alloc((size_t)BATCH * A_MAX * HID * 4);

  // K1: segment mean
  seg_mean_kernel<<<dim3(NWORDS, BATCH), 256, 0, stream>>>(embeddings, mask, words_h);

  // K2: weight conversions to f16
  auto cvt = [&](const float* src, _Float16* dst, int n) {
    cvt_f16_kernel<<<(n + 255) / 256, 256, 0, stream>>>(src, dst, n);
  };
  cvt(w_ih_f, wihf_h, G4 * D_BERT);
  cvt(w_ih_r, wihr_h, G4 * D_BERT);
  cvt(w_hh_f, whhf_h, G4 * HID);
  cvt(w_hh_r, whhr_h, G4 * HID);
  cvt(fc1_w,  fc1w_h, HID * 2 * H2);

  // K3: input-gate GEMM (both dirs), biases folded in
  xg_gemm_kernel<<<dim3(2, BATCH * NWORDS / 16, 2), 256, 0, stream>>>(
      words_h, wihf_h, wihr_h, b_ih_f, b_hh_f, b_ih_r, b_hh_r, xg_f, xg_r);

  // K4: recurrence (dynamic LDS: 16x1024 f32 gates + 16x256 f16 h)
  const size_t lstm_smem = (size_t)16 * G4 * 4 + (size_t)16 * HID * 2;
  lstm_kernel<<<dim3(BATCH / 16, 2), 256, lstm_smem, stream>>>(
      xg_f, xg_r, whhf_h, whhr_h, lstm_o);

  // K5: sentence mean
  sent_mean_kernel<<<(BATCH * H2 + 255) / 256, 256, 0, stream>>>(lstm_o, sent);

  // K6: aspect gather-mean + concat -> f16 emb
  build_emb_kernel<<<(BATCH * A_MAX * 2 * H2 + 255) / 256, 256, 0, stream>>>(
      lstm_o, sent, aidx, emb_h);

  // K7: FC1 (WMMA) + ReLU
  fc1_gemm_kernel<<<dim3(1, 16), 256, 0, stream>>>(emb_h, fc1w_h, fc1_b, h1);

  // K8: FC2 -> logits
  fc2_kernel<<<(BATCH * A_MAX * N_LABELS + 255) / 256, 256, 0, stream>>>(
      h1, fc2_w, fc2_b, out);
}
